// MultiModalFusion_38474317038220
// MI455X (gfx1250) — compile-verified, matrix operand order checked by reading, not run on hardware
//
#include <hip/hip_runtime.h>

// ---------------------------------------------------------------------------
// MultiModalFusion on gfx1250 (MI455X): all GEMMs via v_wmma_f32_16x16x32_f16
// 128x128 block tile, 8 wave32 (4x2), wave tile 32x64 = 8 WMMAs/k-step,
// double-buffered f16 LDS staging with prefetch-before-compute.
// ---------------------------------------------------------------------------

typedef __attribute__((ext_vector_type(16))) _Float16 v16h;
typedef __attribute__((ext_vector_type(8)))  float    v8f;

#define BM 128
#define BN 128
#define BK 32

__device__ __forceinline__ void load_tile16(const float* __restrict__ base,
                                            int stride, int r, int c, int k0,
                                            float4 v[4]) {
    const float* p = base + (size_t)r * stride + (k0 + c);
    v[0] = *(const float4*)(p);
    v[1] = *(const float4*)(p + 4);
    v[2] = *(const float4*)(p + 8);
    v[3] = *(const float4*)(p + 12);
}

__device__ __forceinline__ void store_tile16(_Float16 (*__restrict__ dst)[BK],
                                             int r, int c, const float4 v[4]) {
    alignas(16) _Float16 h16[16] = {
        (_Float16)v[0].x, (_Float16)v[0].y, (_Float16)v[0].z, (_Float16)v[0].w,
        (_Float16)v[1].x, (_Float16)v[1].y, (_Float16)v[1].z, (_Float16)v[1].w,
        (_Float16)v[2].x, (_Float16)v[2].y, (_Float16)v[2].z, (_Float16)v[2].w,
        (_Float16)v[3].x, (_Float16)v[3].y, (_Float16)v[3].z, (_Float16)v[3].w};
    *(uint4*)&dst[r][c]     = ((const uint4*)h16)[0];
    *(uint4*)&dst[r][c + 8] = ((const uint4*)h16)[1];
}

// MODE 0: C[m*ldc+n] = acc + bias[n]
// MODE 1: C[m*ldc+n] = relu(acc + bias[n])
// MODE 2: expert select: e = n>>9, c = n&511; if labels[m]==e: out[m*512+c]=acc+bias[n]
// K compile-time: removes guard branches, enables pipelining.
template <int MODE, int K>
__global__ __launch_bounds__(256) void gemm_wmma_f16(
    const float* __restrict__ A, int lda,          // A: [M,K] fp32
    const float* __restrict__ W,                   // W: [N,K] fp32 (torch out,in)
    const float* __restrict__ bias,                // [N]
    float* __restrict__ C, int ldc,                // [M,ldc]
    const int* __restrict__ labels,                // MODE 2 only
    float* __restrict__ out_sel)                   // MODE 2 only
{
    __shared__ _Float16 Ah[2][BM][BK];   // 16 KB
    __shared__ _Float16 Bh[2][BN][BK];   // 16 KB

    const int tid  = threadIdx.x;
    const int wave = tid >> 5;
    const int lane = tid & 31;
    const int wm   = wave >> 1;       // 0..3  (32-row strip within block)
    const int wn   = wave & 1;        // 0..1  (64-col strip within block)
    const int mrow = lane & 15;       // row (A) / col (B) / col (C) within 16-tile
    const int hh   = lane >> 4;       // lane-half selects K / M sub-groups
    const int rowBase = blockIdx.y * BM;
    const int nBase   = blockIdx.x * BN;

    // cooperative staging coords: 128 rows x 32 cols, 16 floats per thread
    const int sr = tid >> 1;          // 0..127
    const int sc = (tid & 1) * 16;    // 0,16

    v8f acc[2][4] = {};

    // prologue: stage tile k0=0 into buffer 0
    {
        float4 pa[4], pb[4];
        load_tile16(A, lda, rowBase + sr, sc, 0, pa);
        load_tile16(W, K,   nBase  + sr, sc, 0, pb);
        store_tile16(Ah[0], sr, sc, pa);
        store_tile16(Bh[0], sr, sc, pb);
    }
    __syncthreads();

    for (int k0 = 0; k0 < K; k0 += BK) {
        const int buf = (k0 >> 5) & 1;

        // 1) issue next tile's global loads first (overlap with WMMA below)
        float4 pa[4], pb[4];
        if (k0 + BK < K) {
            load_tile16(A, lda, rowBase + sr, sc, k0 + BK, pa);
            load_tile16(W, K,   nBase  + sr, sc, k0 + BK, pb);
        }

        // 2) load ALL fragments for this k-step (one dscnt wait covers them),
        //    then issue the 8 WMMAs back-to-back.
        v16h af[2];
#pragma unroll
        for (int s = 0; s < 2; ++s) {
            const _Float16* ap = &Ah[buf][wm * 32 + s * 16 + mrow][hh * 8];
            ((uint4*)&af[s])[0] = *(const uint4*)ap;
            ((uint4*)&af[s])[1] = *(const uint4*)(ap + 16);
        }
        v16h bf[4];
#pragma unroll
        for (int t = 0; t < 4; ++t) {
            const _Float16* bp = &Bh[buf][wn * 64 + t * 16 + mrow][hh * 8];
            ((uint4*)&bf[t])[0] = *(const uint4*)bp;
            ((uint4*)&bf[t])[1] = *(const uint4*)(bp + 16);
        }
#pragma unroll
        for (int s = 0; s < 2; ++s)
#pragma unroll
            for (int t = 0; t < 4; ++t)
                acc[s][t] = __builtin_amdgcn_wmma_f32_16x16x32_f16(
                    false, af[s], false, bf[t], (short)0, acc[s][t], false, false);

        // 3) convert + store next tile into the other buffer
        if (k0 + BK < K) {
            store_tile16(Ah[buf ^ 1], sr, sc, pa);
            store_tile16(Bh[buf ^ 1], sr, sc, pb);
        }
        __syncthreads();
    }

    // epilogue: C/D f32 16x16 layout: lane 0-15 N=lane, VGPR r -> M=r (+8 hi lanes)
#pragma unroll
    for (int s = 0; s < 2; ++s) {
#pragma unroll
        for (int t = 0; t < 4; ++t) {
            const int n  = nBase + wn * 64 + t * 16 + mrow;
            const float bv = bias[n];
#pragma unroll
            for (int r = 0; r < 8; ++r) {
                const int m = rowBase + wm * 32 + s * 16 + r + hh * 8;
                float v = acc[s][t][r] + bv;
                if constexpr (MODE == 1) v = fmaxf(v, 0.0f);
                if constexpr (MODE == 2) {
                    const int e = n >> 9;
                    const int c = n & 511;
                    if (labels[m] == e) out_sel[(size_t)m * 512 + c] = v;
                } else {
                    C[(size_t)m * ldc + n] = v;
                }
            }
        }
    }
}

// ---------------------------------------------------------------------------
// Attention over seq=3, 8 heads of dim 64. One wave32 per batch element;
// lanes split the 64 head-dims (2 per lane), shfl_xor reductions for scores.
// qkv layout: [B, 3, 1536] with [q|k|v] per token. out: [B*3, 512].
// ---------------------------------------------------------------------------
__global__ __launch_bounds__(256) void attn_seq3_kernel(
    const float* __restrict__ qkv, float* __restrict__ out)
{
    const int wave = threadIdx.x >> 5;
    const int lane = threadIdx.x & 31;
    const int b = blockIdx.x * 8 + wave;
    const float* base = qkv + (size_t)b * 4608;

    for (int h = 0; h < 8; ++h) {
        float q[3][2], k[3][2], v[3][2];
#pragma unroll
        for (int s = 0; s < 3; ++s) {
            const float* p = base + s * 1536 + h * 64;
            q[s][0] = p[lane];        q[s][1] = p[lane + 32];
            k[s][0] = p[512 + lane];  k[s][1] = p[512 + lane + 32];
            v[s][0] = p[1024 + lane]; v[s][1] = p[1024 + lane + 32];
        }
        float sc[3][3];
#pragma unroll
        for (int i = 0; i < 3; ++i)
#pragma unroll
            for (int j = 0; j < 3; ++j) {
                float p = q[i][0] * k[j][0] + q[i][1] * k[j][1];
#pragma unroll
                for (int off = 16; off > 0; off >>= 1)
                    p += __shfl_xor(p, off, 32);
                sc[i][j] = p * 0.125f;      // 1/sqrt(64)
            }
#pragma unroll
        for (int i = 0; i < 3; ++i) {
            float mx = fmaxf(sc[i][0], fmaxf(sc[i][1], sc[i][2]));
            float e0 = __expf(sc[i][0] - mx);
            float e1 = __expf(sc[i][1] - mx);
            float e2 = __expf(sc[i][2] - mx);
            float inv = 1.0f / (e0 + e1 + e2);
            e0 *= inv; e1 *= inv; e2 *= inv;
            float o0 = e0 * v[0][0] + e1 * v[1][0] + e2 * v[2][0];
            float o1 = e0 * v[0][1] + e1 * v[1][1] + e2 * v[2][1];
            float* op = out + ((size_t)b * 3 + i) * 512 + h * 64;
            op[lane]      = o0;
            op[lane + 32] = o1;
        }
    }
}

// ---------------------------------------------------------------------------
// LayerNorm over E=512, in place. One wave32 per row, 16 elems per lane.
// ---------------------------------------------------------------------------
__global__ __launch_bounds__(256) void layernorm512_kernel(
    float* __restrict__ y, const float* __restrict__ g, const float* __restrict__ b)
{
    const int wave = threadIdx.x >> 5;
    const int lane = threadIdx.x & 31;
    const int row = blockIdx.x * 8 + wave;
    float* p = y + (size_t)row * 512;

    float vals[16];
    float s = 0.0f, s2 = 0.0f;
#pragma unroll
    for (int i = 0; i < 16; ++i) {
        float v = p[lane + i * 32];
        vals[i] = v; s += v; s2 += v * v;
    }
#pragma unroll
    for (int off = 16; off > 0; off >>= 1) {
        s  += __shfl_xor(s,  off, 32);
        s2 += __shfl_xor(s2, off, 32);
    }
    const float mu  = s * (1.0f / 512.0f);
    const float var = s2 * (1.0f / 512.0f) - mu * mu;
    const float rs  = __frsqrt_rn(var + 1e-5f);
#pragma unroll
    for (int i = 0; i < 16; ++i) {
        const int c = lane + i * 32;
        p[c] = (vals[i] - mu) * rs * g[c] + b[c];
    }
}

// ---------------------------------------------------------------------------
extern "C" void kernel_launch(void* const* d_in, const int* in_sizes, int n_in,
                              void* d_out, int out_size, void* d_ws, size_t ws_size,
                              hipStream_t stream) {
    (void)in_sizes; (void)n_in; (void)out_size; (void)ws_size;
    constexpr int Bsz = 16384;

    const float* img   = (const float*)d_in[0];
    const float* txt   = (const float*)d_in[1];
    const float* knw   = (const float*)d_in[2];
    const int*   lbl   = (const int*)d_in[3];
    const float* ipw   = (const float*)d_in[4];   // [1536,512]
    const float* ipb   = (const float*)d_in[5];
    const float* opw   = (const float*)d_in[6];   // [512,512]
    const float* opb   = (const float*)d_in[7];
    const float* fw1   = (const float*)d_in[8];   // [256,1536]
    const float* fb1   = (const float*)d_in[9];
    const float* fw2   = (const float*)d_in[10];  // [512,256]
    const float* fb2   = (const float*)d_in[11];
    const float* lng   = (const float*)d_in[12];
    const float* lnb   = (const float*)d_in[13];
    const float* affw  = (const float*)d_in[14];  // [4,512,512] -> [2048,512]
    const float* affb  = (const float*)d_in[15];  // [4,512]     -> [2048]

    // Workspace layout (buffers recycled; peak ~403 MB):
    //   region0 [0, 302MB):       qkv [B,3,1536]   -> later fus_in [3B,512]
    //   region1 [302MB, 403MB):   attn_out [3B,512] -> later h1 [B,256], y [B,512]
    char* ws = (char*)d_ws;
    const size_t qkv_bytes = (size_t)Bsz * 3 * 1536 * sizeof(float);
    float* qkv      = (float*)ws;
    float* attn_out = (float*)(ws + qkv_bytes);
    float* fus_in   = (float*)ws;                                        // reuse qkv
    float* h1       = (float*)(ws + qkv_bytes);                          // reuse attn_out
    float* yv       = (float*)(ws + qkv_bytes + (size_t)Bsz * 256 * 4);  // after h1

    const dim3 blk(256);

    // 1) QKV projection: one launch per modality, M=16384, N=1536, K=512
    const float* srcs[3] = {img, txt, knw};
    for (int s = 0; s < 3; ++s) {
        gemm_wmma_f16<0, 512><<<dim3(1536 / BN, Bsz / BM), blk, 0, stream>>>(
            srcs[s], 512, ipw, ipb, qkv + s * 1536, 4608, nullptr, nullptr);
    }

    // 2) attention (seq=3), one wave per batch element
    attn_seq3_kernel<<<Bsz / 8, blk, 0, stream>>>(qkv, attn_out);

    // 3) out_proj: M=3B=49152, N=512, K=512; output = fusion input [B,1536]
    gemm_wmma_f16<0, 512><<<dim3(512 / BN, 3 * Bsz / BM), blk, 0, stream>>>(
        attn_out, 512, opw, opb, fus_in, 512, nullptr, nullptr);

    // 4) fusion layer 1 + ReLU: M=16384, N=256, K=1536
    gemm_wmma_f16<1, 1536><<<dim3(256 / BN, Bsz / BM), blk, 0, stream>>>(
        fus_in, 1536, fw1, fb1, h1, 256, nullptr, nullptr);

    // 5) fusion layer 2: M=16384, N=512, K=256
    gemm_wmma_f16<0, 256><<<dim3(512 / BN, Bsz / BM), blk, 0, stream>>>(
        h1, 256, fw2, fb2, yv, 512, nullptr, nullptr);

    // 6) LayerNorm(512), in place
    layernorm512_kernel<<<Bsz / 8, blk, 0, stream>>>(yv, lng, lnb);

    // 7) all 4 experts as one N=2048 GEMM with per-row expert-select epilogue
    gemm_wmma_f16<2, 512><<<dim3(2048 / BN, Bsz / BM), blk, 0, stream>>>(
        yv, 512, affw, affb, nullptr, 0, lbl, (float*)d_out);
}
